// Gloabal_MultiHeadAttention_14989435863425
// MI455X (gfx1250) — compile-verified, matrix operand order checked by reading, not run on hardware
//
#include <hip/hip_runtime.h>
#include <hip/hip_bf16.h>

typedef __attribute__((ext_vector_type(16))) _Float16 v16h;
typedef __attribute__((ext_vector_type(8)))  float    v8f;

namespace {

constexpr int   Bb   = 4;
constexpr int   Nn   = 2048;
constexpr int   Cc   = 1024;
constexpr int   Hh   = 16;
constexpr int   Dd   = 64;
constexpr float NEGV = -1e9f;

union AFrag { v16h v; uint4 q[2]; };

__device__ __forceinline__ v8f wmma16(const AFrag& a, const AFrag& b, v8f c) {
  // D = A(16x32 f16) * B(32x16 f16) + C(16x16 f32)
  return __builtin_amdgcn_wmma_f32_16x16x32_f16(
      /*neg_a=*/false, a.v, /*neg_b=*/false, b.v,
      /*c_mod=*/(short)0, c, /*reuse_a=*/false, /*reuse_b=*/false);
}

// ---------------------------------------------------------------- cast f32->f16
__global__ void cast_f16_kernel(const float* __restrict__ in,
                                _Float16* __restrict__ out) {
  int i = blockIdx.x * blockDim.x + threadIdx.x;  // one float4 per thread
  float4 f = ((const float4*)in)[i];
  __align__(8) _Float16 h[4] = {(_Float16)f.x, (_Float16)f.y,
                                (_Float16)f.z, (_Float16)f.w};
  ((uint2*)out)[i] = *(const uint2*)h;
}

// -------------------------------------------- transpose+cast weight: WT[n][k]
__global__ void transcast_kernel(const float* __restrict__ W,
                                 _Float16* __restrict__ WT) {
  __shared__ float tile[32][33];
  int bx = blockIdx.x, by = blockIdx.y;
  int tx = threadIdx.x, ty = threadIdx.y;
#pragma unroll
  for (int yy = 0; yy < 32; yy += 8)
    tile[ty + yy][tx] = W[(size_t)(by * 32 + ty + yy) * Cc + bx * 32 + tx];
  __syncthreads();
#pragma unroll
  for (int yy = 0; yy < 32; yy += 8)
    WT[(size_t)(bx * 32 + ty + yy) * Cc + by * 32 + tx] =
        (_Float16)tile[tx][ty + yy];
}

// ------------------------------------------------------------------ main GEMM
// Y[M=8192, 1024] = A[M,1024](f16) @ BT[1024,1024](f16, K-contiguous)
// mode 0: write f16 to [B,H,N,D] layout, value scaled by `scale`
// mode 1: write f32  residual[r,c] + val  to outF[r,c]  ([B*N, C])
__global__ void __launch_bounds__(256) gemm_wmma_kernel(
    const _Float16* __restrict__ A, const _Float16* __restrict__ BT,
    _Float16* __restrict__ outH, float* __restrict__ outF,
    const float* __restrict__ residual, float scale, int mode) {
  __shared__ _Float16 As[128 * 32];
  __shared__ _Float16 Bs[128 * 32];
  const int t = threadIdx.x;
  const int lane = t & 31, w = t >> 5;
  const int r = lane & 15, hi = lane >> 4;
  const int wm = w >> 2, wn = w & 3;             // waves: 2 (M) x 4 (N)
  const int bm = blockIdx.x * 128, bn = blockIdx.y * 128;
  const int lr = t >> 1, ls = (t & 1) << 4;      // 2 threads per 32-half row

  v8f z = {};
  v8f acc[4][2];
#pragma unroll
  for (int mt = 0; mt < 4; ++mt)
#pragma unroll
    for (int nt = 0; nt < 2; ++nt) acc[mt][nt] = z;

  for (int k0 = 0; k0 < Cc; k0 += 32) {
    __syncthreads();
    {
      const uint4* ga = (const uint4*)(A + (size_t)(bm + lr) * Cc + k0 + ls);
      uint4* sa = (uint4*)(As + lr * 32 + ls);
      sa[0] = ga[0]; sa[1] = ga[1];
      const uint4* gb = (const uint4*)(BT + (size_t)(bn + lr) * Cc + k0 + ls);
      uint4* sb = (uint4*)(Bs + lr * 32 + ls);
      sb[0] = gb[0]; sb[1] = gb[1];
    }
    __syncthreads();
    AFrag af[4], bf[2];
#pragma unroll
    for (int mt = 0; mt < 4; ++mt) {
      const _Float16* p = As + (wm * 64 + mt * 16 + r) * 32;
      af[mt].q[0] = *(const uint4*)(p + hi * 8);
      af[mt].q[1] = *(const uint4*)(p + 16 + hi * 8);
    }
#pragma unroll
    for (int nt = 0; nt < 2; ++nt) {
      const _Float16* p = Bs + (wn * 32 + nt * 16 + r) * 32;
      bf[nt].q[0] = *(const uint4*)(p + hi * 8);
      bf[nt].q[1] = *(const uint4*)(p + 16 + hi * 8);
    }
#pragma unroll
    for (int mt = 0; mt < 4; ++mt)
#pragma unroll
      for (int nt = 0; nt < 2; ++nt)
        acc[mt][nt] = wmma16(af[mt], bf[nt], acc[mt][nt]);
  }

#pragma unroll
  for (int mt = 0; mt < 4; ++mt)
#pragma unroll
    for (int nt = 0; nt < 2; ++nt)
#pragma unroll
      for (int v = 0; v < 8; ++v) {
        int gr = bm + wm * 64 + mt * 16 + hi * 8 + v;
        int gc = bn + wn * 32 + nt * 16 + r;
        float val = acc[mt][nt][v] * scale;
        if (mode == 0) {
          int b = gr >> 11, i = gr & (Nn - 1);
          int h = gc >> 6, d = gc & 63;
          outH[(((size_t)(b * Hh + h)) * Nn + i) * Dd + d] = (_Float16)val;
        } else {
          size_t o = (size_t)gr * Cc + gc;
          outF[o] = residual[o] + val;
        }
      }
}

// ------------------------------------- pass 1: per-(b,h,j) column max / sumexp
// s[i,j] = (mask ? NEG : qh_i . kh_j) * ratio[b,j]   (scale folded into qh)
__global__ void __launch_bounds__(256) stats_kernel(
    const _Float16* __restrict__ qh, const _Float16* __restrict__ kh,
    const unsigned char* __restrict__ mask, const float* __restrict__ ratio,
    float* __restrict__ Mbuf, float* __restrict__ Zbuf) {
  __shared__ _Float16 Kt[128 * 64];
  __shared__ _Float16 Qs[128 * 64];
  int bx = blockIdx.x;
  int jb = bx & 15, bh = bx >> 4;
  int b = bh >> 4;
  int jbase = jb * 128;
  int t = threadIdx.x, w = t >> 5, lane = t & 31, r = lane & 15, hi = lane >> 4;
  {
    int jj = t >> 1, seg = (t & 1) * 32;
    const uint4* g = (const uint4*)(kh + ((size_t)bh * Nn + jbase + jj) * Dd + seg);
    uint4* s = (uint4*)(Kt + jj * 64 + seg);
    s[0] = g[0]; s[1] = g[1]; s[2] = g[2]; s[3] = g[3];
  }
  __syncthreads();
  AFrag bk[2];
#pragma unroll
  for (int kk = 0; kk < 2; ++kk) {
    const _Float16* p = Kt + (w * 16 + r) * 64 + kk * 32;
    bk[kk].q[0] = *(const uint4*)(p + hi * 8);
    bk[kk].q[1] = *(const uint4*)(p + 16 + hi * 8);
  }
  int j = jbase + w * 16 + r;                 // this lane's column
  float ratio_j = ratio[b * Nn + j];
  float Mrun = -3.0e38f, Zrun = 0.f;

  for (int ib = 0; ib < Nn; ib += 128) {
    __syncthreads();
    {
      int ii = t >> 1, seg = (t & 1) * 32;
      const uint4* g = (const uint4*)(qh + ((size_t)bh * Nn + ib + ii) * Dd + seg);
      uint4* s = (uint4*)(Qs + ii * 64 + seg);
      s[0] = g[0]; s[1] = g[1]; s[2] = g[2]; s[3] = g[3];
    }
    __syncthreads();
    for (int it = 0; it < 8; ++it) {
      AFrag aq[2];
#pragma unroll
      for (int kk = 0; kk < 2; ++kk) {
        const _Float16* p = Qs + (it * 16 + r) * 64 + kk * 32;
        aq[kk].q[0] = *(const uint4*)(p + hi * 8);
        aq[kk].q[1] = *(const uint4*)(p + 16 + hi * 8);
      }
      v8f acc = {};
      acc = wmma16(aq[0], bk[0], acc);
      acc = wmma16(aq[1], bk[1], acc);
      int i0 = ib + it * 16 + hi * 8;
      const unsigned char* mp = mask + ((size_t)b * Nn + i0) * Nn + j;
      float s8[8];
#pragma unroll
      for (int v = 0; v < 8; ++v) {
        float sv = acc[v];
        sv = mp[(size_t)v * Nn] ? NEGV : sv;
        s8[v] = sv * ratio_j;
      }
      float mt = s8[0];
#pragma unroll
      for (int v = 1; v < 8; ++v) mt = fmaxf(mt, s8[v]);
      float Mn = fmaxf(Mrun, mt);
      float zs = __expf(Mrun - Mn);
      float za = 0.f;
#pragma unroll
      for (int v = 0; v < 8; ++v) za += __expf(s8[v] - Mn);
      Zrun = Zrun * zs + za;
      Mrun = Mn;
    }
  }
  float Mo = __shfl_xor(Mrun, 16);
  float Zo = __shfl_xor(Zrun, 16);
  float Mc = fmaxf(Mrun, Mo);
  float Zc = Zrun * __expf(Mrun - Mc) + Zo * __expf(Mo - Mc);
  if (hi == 0) {
    Mbuf[(size_t)bh * Nn + j] = Mc;
    Zbuf[(size_t)bh * Nn + j] = Zc;
  }
}

// --------------------------------------------------- fold 1/Z[j] into v -> v2
__global__ void fold_kernel(const _Float16* __restrict__ vh,
                            const float* __restrict__ Z,
                            _Float16* __restrict__ v2) {
  int i = blockIdx.x * blockDim.x + threadIdx.x;   // [B,H,N,D] flat
  v2[i] = (_Float16)((float)vh[i] * (1.0f / Z[i >> 6]));
}

// ------------- pass 2: ctx[i,d] = sum_j exp(s[i,j]-M[j]) * v2[j,d]  (WMMA x2)
__global__ void __launch_bounds__(256) ctx_kernel(
    const _Float16* __restrict__ qh, const _Float16* __restrict__ kh,
    const _Float16* __restrict__ v2, const unsigned char* __restrict__ mask,
    const float* __restrict__ ratio, const float* __restrict__ Mbuf,
    _Float16* __restrict__ ctx) {
  __shared__ __align__(16) char smem[50176];
  _Float16* Kt  = (_Float16*)smem;                // 128 x 64   (16 KB)
  _Float16* Vt  = (_Float16*)(smem + 16384);      // 64 x 128   (16 KB), [d][j]
  _Float16* Wst = (_Float16*)(smem + 32768);      // 8 waves x 16 x 64 (16 KB);
                                                  // doubles as Qs at startup
  float* Ms = (float*)(smem + 49152);
  float* Rs = (float*)(smem + 49664);

  int bx = blockIdx.x;
  int ibt = bx & 15, bh = bx >> 4, b = bh >> 4, h = bh & 15;
  int ibase = ibt * 128;
  int t = threadIdx.x, w = t >> 5, lane = t & 31, r = lane & 15, hi = lane >> 4;

  {  // stage Q block once (into Wst region, consumed into registers below)
    int ii = t >> 1, seg = (t & 1) * 32;
    const uint4* g = (const uint4*)(qh + ((size_t)bh * Nn + ibase + ii) * Dd + seg);
    uint4* s = (uint4*)(Wst + ii * 64 + seg);
    s[0] = g[0]; s[1] = g[1]; s[2] = g[2]; s[3] = g[3];
  }
  __syncthreads();
  AFrag aq[2];
#pragma unroll
  for (int kk = 0; kk < 2; ++kk) {
    const _Float16* p = Wst + (w * 16 + r) * 64 + kk * 32;
    aq[kk].q[0] = *(const uint4*)(p + hi * 8);
    aq[kk].q[1] = *(const uint4*)(p + 16 + hi * 8);
  }
  v8f z = {};
  v8f acc[4];
#pragma unroll
  for (int dt = 0; dt < 4; ++dt) acc[dt] = z;

  _Float16* wrow = Wst + w * 1024;   // this wave's 16x64 staging area

  for (int jb = 0; jb < Nn; jb += 128) {
    __syncthreads();
    {
      int jj = t >> 1, seg = (t & 1) * 32;
      const uint4* gk = (const uint4*)(kh + ((size_t)bh * Nn + jb + jj) * Dd + seg);
      uint4* sk = (uint4*)(Kt + jj * 64 + seg);
      sk[0] = gk[0]; sk[1] = gk[1]; sk[2] = gk[2]; sk[3] = gk[3];
      __align__(16) _Float16 tv[32];
      const uint4* gv = (const uint4*)(v2 + ((size_t)bh * Nn + jb + jj) * Dd + seg);
      *(uint4*)(tv + 0)  = gv[0]; *(uint4*)(tv + 8)  = gv[1];
      *(uint4*)(tv + 16) = gv[2]; *(uint4*)(tv + 24) = gv[3];
#pragma unroll
      for (int dd = 0; dd < 32; ++dd) Vt[(seg + dd) * 128 + jj] = tv[dd];
    }
    if (t < 128) {
      Ms[t] = Mbuf[(size_t)bh * Nn + jb + t];
      Rs[t] = ratio[(size_t)b * Nn + jb + t];
    }
    __syncthreads();

#pragma unroll
    for (int half = 0; half < 2; ++half) {
      // stage 1: w-tile 16 x 64 (4 j-tiles) via WMMA + exp epilogue -> Wst
      for (int jt4 = 0; jt4 < 4; ++jt4) {
        int jt = half * 4 + jt4;
        AFrag bk[2];
#pragma unroll
        for (int kk = 0; kk < 2; ++kk) {
          const _Float16* p = Kt + (jt * 16 + r) * 64 + kk * 32;
          bk[kk].q[0] = *(const uint4*)(p + hi * 8);
          bk[kk].q[1] = *(const uint4*)(p + 16 + hi * 8);
        }
        v8f s = {};
        s = wmma16(aq[0], bk[0], s);
        s = wmma16(aq[1], bk[1], s);
        int jloc = jt * 16 + r;
        float Mj = Ms[jloc], rt = Rs[jloc];
        int i0 = ibase + w * 16 + hi * 8;
        const unsigned char* mp = mask + ((size_t)b * Nn + i0) * Nn + (jb + jloc);
#pragma unroll
        for (int v = 0; v < 8; ++v) {
          float sv = s[v];
          sv = mp[(size_t)v * Nn] ? NEGV : sv;
          sv *= rt;
          wrow[(hi * 8 + v) * 64 + jt4 * 16 + r] = (_Float16)__expf(sv - Mj);
        }
      }
      // stage 2: acc += W(16x64) @ V2t(64x64)
#pragma unroll
      for (int kk = 0; kk < 2; ++kk) {
        AFrag aw;
        const _Float16* p = wrow + r * 64 + kk * 32;
        aw.q[0] = *(const uint4*)(p + hi * 8);
        aw.q[1] = *(const uint4*)(p + 16 + hi * 8);
#pragma unroll
        for (int dt = 0; dt < 4; ++dt) {
          AFrag bv;
          const _Float16* pv = Vt + (dt * 16 + r) * 128 + half * 64 + kk * 32;
          bv.q[0] = *(const uint4*)(pv + hi * 8);
          bv.q[1] = *(const uint4*)(pv + 16 + hi * 8);
          acc[dt] = wmma16(aw, bv, acc[dt]);
        }
      }
    }
  }
  // ctx layout [B,N,C] f16, c = h*64 + d
#pragma unroll
  for (int dt = 0; dt < 4; ++dt)
#pragma unroll
    for (int v = 0; v < 8; ++v) {
      int i = ibase + w * 16 + hi * 8 + v;
      ctx[((size_t)b * Nn + i) * Cc + h * 64 + dt * 16 + r] = (_Float16)acc[dt][v];
    }
}

// --------------------------------------------------------- in-place LayerNorm
__global__ void __launch_bounds__(256) layernorm_kernel(
    float* __restrict__ x, const float* __restrict__ gamma,
    const float* __restrict__ beta) {
  __shared__ float red[16];
  __shared__ float stat[2];
  int row = blockIdx.x, t = threadIdx.x;
  float4 v = ((const float4*)(x + (size_t)row * Cc))[t];
  float s = v.x + v.y + v.z + v.w;
  float ss = v.x * v.x + v.y * v.y + v.z * v.z + v.w * v.w;
#pragma unroll
  for (int o = 16; o > 0; o >>= 1) {
    s += __shfl_xor(s, o);
    ss += __shfl_xor(ss, o);
  }
  int w = t >> 5, lane = t & 31;
  if (lane == 0) { red[w] = s; red[8 + w] = ss; }
  __syncthreads();
  if (t == 0) {
    float S = 0.f, SS = 0.f;
#pragma unroll
    for (int i = 0; i < 8; ++i) { S += red[i]; SS += red[8 + i]; }
    float mu = S * (1.0f / Cc);
    float var = SS * (1.0f / Cc) - mu * mu;
    stat[0] = mu;
    stat[1] = rsqrtf(var + 1e-5f);
  }
  __syncthreads();
  float mu = stat[0], rs = stat[1];
  float4 g = ((const float4*)gamma)[t];
  float4 be = ((const float4*)beta)[t];
  float4 o;
  o.x = (v.x - mu) * rs * g.x + be.x;
  o.y = (v.y - mu) * rs * g.y + be.y;
  o.z = (v.z - mu) * rs * g.z + be.z;
  o.w = (v.w - mu) * rs * g.w + be.w;
  ((float4*)(x + (size_t)row * Cc))[t] = o;
}

}  // namespace

extern "C" void kernel_launch(void* const* d_in, const int* in_sizes, int n_in,
                              void* d_out, int out_size, void* d_ws,
                              size_t ws_size, hipStream_t stream) {
  (void)in_sizes; (void)n_in; (void)out_size; (void)ws_size;
  const float* query = (const float*)d_in[0];
  const float* key_  = (const float*)d_in[1];
  const float* value = (const float*)d_in[2];
  const float* ratio = (const float*)d_in[3];
  const unsigned char* amask = (const unsigned char*)d_in[4];  // jnp.bool_ = 1B
  const float* Wq = (const float*)d_in[5];
  const float* Wk = (const float*)d_in[6];
  const float* Wv = (const float*)d_in[7];
  const float* Wo = (const float*)d_in[8];
  const float* gamma = (const float*)d_in[9];
  const float* beta  = (const float*)d_in[10];
  float* out = (float*)d_out;

  const size_t MC  = (size_t)Bb * Nn * Cc;       // 8M elements
  const size_t WC  = (size_t)Cc * Cc;            // 1M elements
  char* ws = (char*)d_ws;
  size_t off = 0;
  auto carve = [&](size_t bytes) -> char* {
    char* p = ws + off;
    off += (bytes + 255) & ~(size_t)255;
    return p;
  };
  _Float16* xqh = (_Float16*)carve(MC * 2);
  _Float16* xkh = (_Float16*)carve(MC * 2);
  _Float16* xvh = (_Float16*)carve(MC * 2);
  _Float16* WqT = (_Float16*)carve(WC * 2);
  _Float16* WkT = (_Float16*)carve(WC * 2);
  _Float16* WvT = (_Float16*)carve(WC * 2);
  _Float16* WoT = (_Float16*)carve(WC * 2);
  _Float16* qh  = (_Float16*)carve(MC * 2);
  _Float16* kh  = (_Float16*)carve(MC * 2);
  _Float16* vh  = (_Float16*)carve(MC * 2);
  float* Mbuf   = (float*)carve((size_t)Bb * Hh * Nn * 4);
  float* Zbuf   = (float*)carve((size_t)Bb * Hh * Nn * 4);
  _Float16* v2  = xqh;   // alias: xqh dead after Q projection
  _Float16* ctx = xkh;   // alias: xkh dead after K projection

  const float qscale = 0.125f;  // D^-0.5 = 1/8, folded into Q projection

  // 1) cast activations to f16
  cast_f16_kernel<<<MC / (256 * 4), 256, 0, stream>>>(query, xqh);
  cast_f16_kernel<<<MC / (256 * 4), 256, 0, stream>>>(key_, xkh);
  cast_f16_kernel<<<MC / (256 * 4), 256, 0, stream>>>(value, xvh);
  // 2) transpose+cast weights (K-contiguous for WMMA B operand)
  dim3 tg(32, 32), tb(32, 8);
  transcast_kernel<<<tg, tb, 0, stream>>>(Wq, WqT);
  transcast_kernel<<<tg, tb, 0, stream>>>(Wk, WkT);
  transcast_kernel<<<tg, tb, 0, stream>>>(Wv, WvT);
  transcast_kernel<<<tg, tb, 0, stream>>>(Wo, WoT);
  // 3) projections -> [B,H,N,D] f16 (Q pre-scaled)
  dim3 gg(Bb * Nn / 128, Cc / 128);
  gemm_wmma_kernel<<<gg, 256, 0, stream>>>(xqh, WqT, qh, nullptr, nullptr, qscale, 0);
  gemm_wmma_kernel<<<gg, 256, 0, stream>>>(xkh, WkT, kh, nullptr, nullptr, 1.0f, 0);
  gemm_wmma_kernel<<<gg, 256, 0, stream>>>(xvh, WvT, vh, nullptr, nullptr, 1.0f, 0);
  // 4) column (query-axis) softmax stats M[j], Z[j]
  stats_kernel<<<Bb * Hh * 16, 256, 0, stream>>>(qh, kh, amask, ratio, Mbuf, Zbuf);
  // 5) fold 1/Z into V
  fold_kernel<<<(Bb * Hh * Nn * Dd) / 256, 256, 0, stream>>>(vh, Zbuf, v2);
  // 6) ctx = exp(s - M[j]) @ v2  -> [B,N,C] f16
  ctx_kernel<<<Bb * Hh * 16, 256, 0, stream>>>(qh, kh, v2, amask, ratio, Mbuf, ctx);
  // 7) out = residual + ctx @ Wo  (f32, straight into d_out)
  gemm_wmma_kernel<<<gg, 256, 0, stream>>>(ctx, WoT, nullptr, out, query, 1.0f, 1);
  // 8) LayerNorm in place on d_out
  layernorm_kernel<<<Bb * Nn, 256, 0, stream>>>(out, gamma, beta);
}
